// Attention_62294205661658
// MI455X (gfx1250) — matrix operand order repaired and checked by reading, not compiled, not run on hardware
//
#include <hip/hip_runtime.h>
#include <hip/hip_bf16.h>

// ---------------------------------------------------------------------------
// Fused MHA forward for MI455X (gfx1250, wave32, WMMA, async-to-LDS).
//   q = x@Wq^T + bq (scaled by SCALE*log2e) -> bf16 [b,h,n,d]
//   k = x@Wk^T + bk                         -> bf16 [b,h,n,d]
//   v = x@Wv^T + bv                         -> bf16 [b,h,d,n]  (TRANSPOSED)
//   flash attention (S^T = K Q^T, O^T = V^T P^T, per-lane online softmax),
//     K / V^T tiles double-buffered in LDS via global_load_async_to_lds_b128
//   out = attn @ Wo^T + bo  (fp32 output)
// Workspace: 4 * 8192*1024 bf16 = 64 MB.
// ---------------------------------------------------------------------------

#define DIM      1024
#define NHEADS   16
#define HEAD_DIM 64
#define BATCH    4
#define SEQ      2048
#define ROWS     (BATCH * SEQ)          // 8192
#define SCALE_F  0.125f                 // HEAD_DIM^-0.5
#define LOG2E_F  1.4426950408889634f

typedef __bf16 bf16x16 __attribute__((ext_vector_type(16)));
typedef float  f32x8   __attribute__((ext_vector_type(8)));
typedef unsigned short u16;
typedef unsigned int   u32;
typedef unsigned long long u64;

union ABOp { bf16x16 h; u32 u[8]; };

static __device__ __forceinline__ u16 f2bf(float f) {
  u32 u = __float_as_uint(f);
  u += 0x7fffu + ((u >> 16) & 1u);     // round to nearest even
  return (u16)(u >> 16);
}
static __device__ __forceinline__ u32 pkbf(float a, float b) {
  return (u32)f2bf(a) | ((u32)f2bf(b) << 16);
}
static __device__ __forceinline__ f32x8 wmma_bf16(const ABOp& a, const ABOp& b, f32x8 c) {
  return __builtin_amdgcn_wmma_f32_16x16x32_bf16(false, a.h, false, b.h,
                                                 (short)0, c, false, false);
}
static __device__ __forceinline__ f32x8 zero8() {
  f32x8 z = {0.f, 0.f, 0.f, 0.f, 0.f, 0.f, 0.f, 0.f};
  return z;
}

// LDS byte offset of a generic pointer into __shared__ (LDS flat addr low bits)
static __device__ __forceinline__ u32 lds_off(const void* p) {
  return (u32)(u64)p;
}
// async DMA: 16 bytes per lane, global -> LDS, tracked by ASYNCcnt
static __device__ __forceinline__ void async_ld_b128(u32 lds_addr, const void* gaddr) {
  asm volatile("global_load_async_to_lds_b128 %0, %1, off"
               :: "v"(lds_addr), "v"((u64)gaddr) : "memory");
}
static __device__ __forceinline__ void wait_async0() {
  asm volatile("s_wait_asynccnt 0x0" ::: "memory");
}

// ---------------------------------------------------------------------------
// QKV projection: Y = X * W^T + b.
// q/k stored bf16 [b,h,n,d]; v stored bf16 [b,h,d,n] (transposed for flash).
// Block 256 thr (8 waves): tile M=128, N=64; wave w does rows 16w..16w+15.
// ---------------------------------------------------------------------------
__global__ __launch_bounds__(256)
void proj_qkv_kernel(const float* __restrict__ xq, const float* __restrict__ xk,
                     const float* __restrict__ xv,
                     const float* __restrict__ Wq, const float* __restrict__ Wk,
                     const float* __restrict__ Wv,
                     const float* __restrict__ bq, const float* __restrict__ bk,
                     const float* __restrict__ bv,
                     u16* __restrict__ oq, u16* __restrict__ ok, u16* __restrict__ ov)
{
  const int z = blockIdx.z;
  const float* X    = (z == 0) ? xq : (z == 1) ? xk : xv;
  const float* W    = (z == 0) ? Wq : (z == 1) ? Wk : Wv;
  const float* bias = (z == 0) ? bq : (z == 1) ? bk : bv;
  u16*         out  = (z == 0) ? oq : (z == 1) ? ok : ov;
  const float  sc   = (z == 0) ? (SCALE_F * LOG2E_F) : 1.0f;

  __shared__ __align__(16) u16 sA[128 * 40];   // X tile, row stride 40 halves
  __shared__ __align__(16) u16 sB[64 * 40];    // W tile

  const int t    = threadIdx.x;
  const int wave = t >> 5;
  const int lane = t & 31;
  const int half = lane >> 4;
  const int l    = lane & 15;
  const int mbase = blockIdx.x * 128;
  const int cbase = blockIdx.y * 64;

  f32x8 acc[4];
#pragma unroll
  for (int i = 0; i < 4; ++i) acc[i] = zero8();

  for (int kb = 0; kb < DIM; kb += 32) {
    // cooperative: X tile 128x32 fp32 -> bf16 LDS
#pragma unroll
    for (int i = 0; i < 4; ++i) {
      int c = t + i * 256;                       // 1024 float4 chunks
      int row = c >> 3, f4 = c & 7;
      float4 v = *(const float4*)&X[(mbase + row) * DIM + kb + f4 * 4];
      *(uint2*)&sA[row * 40 + f4 * 4] = make_uint2(pkbf(v.x, v.y), pkbf(v.z, v.w));
    }
    // cooperative: W tile 64x32 fp32 -> bf16 LDS (rows n, contiguous k)
#pragma unroll
    for (int i = 0; i < 2; ++i) {
      int c = t + i * 256;                       // 512 float4 chunks
      int row = c >> 3, f4 = c & 7;
      float4 v = *(const float4*)&W[(cbase + row) * DIM + kb + f4 * 4];
      *(uint2*)&sB[row * 40 + f4 * 4] = make_uint2(pkbf(v.x, v.y), pkbf(v.z, v.w));
    }
    __syncthreads();

    ABOp a;
#pragma unroll
    for (int v = 0; v < 8; ++v) {
      int k0 = half * 16 + v * 2;     // contiguous k order: k = 16*half + 2v (+1)
      a.u[v] = *(const u32*)&sA[(wave * 16 + l) * 40 + k0];
    }
#pragma unroll
    for (int nt = 0; nt < 4; ++nt) {
      ABOp b;
#pragma unroll
      for (int v = 0; v < 8; ++v)                           // same contiguous k order as A
        b.u[v] = *(const u32*)&sB[(nt * 16 + l) * 40 + half * 16 + v * 2];
      acc[nt] = wmma_bf16(a, b, acc[nt]);
    }
    __syncthreads();
  }

  // store bf16 (C layout: N = lane&15, M = r + 8*half)
#pragma unroll
  for (int nt = 0; nt < 4; ++nt) {
    int cg = cbase + nt * 16 + l;
    int h = cg >> 6, d = cg & 63;
    float bsv = bias[cg];
    if (z != 2) {                       // q/k: [b, h, n, d]
#pragma unroll
      for (int r = 0; r < 8; ++r) {
        int grow = mbase + wave * 16 + r + 8 * half;
        int bb = grow >> 11, n = grow & 2047;
        float val = (acc[nt][r] + bsv) * sc;
        out[(((size_t)(bb * NHEADS + h) * SEQ) + n) * HEAD_DIM + d] = f2bf(val);
      }
    } else {                            // v: [b, h, d, n] transposed
      int g0 = mbase + wave * 16 + 8 * half;   // 8 consecutive n per lane
      int bb = g0 >> 11, n0 = g0 & 2047;
      float vv[8];
#pragma unroll
      for (int r = 0; r < 8; ++r) vv[r] = acc[nt][r] + bsv;
      uint4 pk = make_uint4(pkbf(vv[0], vv[1]), pkbf(vv[2], vv[3]),
                            pkbf(vv[4], vv[5]), pkbf(vv[6], vv[7]));
      *(uint4*)&out[((size_t)(bb * NHEADS + h) * HEAD_DIM + d) * SEQ + n0] = pk;
    }
  }
}

// ---------------------------------------------------------------------------
// Flash attention. Grid (SEQ/128, B*H). Wave handles 16 queries x full d=64.
// Computes S^T = K Q^T (lanes = queries) so softmax is per-lane, then
// O^T += V^T P^T with one cross-half shuffle to build the P^T operand.
// K and V^T tiles stream through double-buffered LDS via async-to-LDS DMA.
// Q was pre-scaled by SCALE*log2e, so exp2f gives the softmax directly.
// ---------------------------------------------------------------------------
__global__ __launch_bounds__(256)
void flash_attn_kernel(const u16* __restrict__ qg, const u16* __restrict__ kg,
                       const u16* __restrict__ vtg, u16* __restrict__ og)
{
  __shared__ __align__(16) u16 sK[2][32 * 72];    // K tile [key][d], stride 72
  __shared__ __align__(16) u16 sVT[2][64 * 40];   // V^T tile [d][key], stride 40

  const int t    = threadIdx.x;
  const int wave = t >> 5;
  const int lane = t & 31;
  const int half = lane >> 4;
  const int l    = lane & 15;
  const int bh   = blockIdx.y;
  const int qrow = blockIdx.x * 128 + wave * 16;

  const u16* qptr  = qg  + (size_t)bh * SEQ * HEAD_DIM;
  const u16* kptr  = kg  + (size_t)bh * SEQ * HEAD_DIM;
  const u16* vtptr = vtg + (size_t)bh * HEAD_DIM * SEQ;   // [d][n]

  // issue one 32-key tile: 256 thr x 2 x 16B = K(4KB) + V^T(4KB), no VGPR data
  auto issue_tile = [&](int buf, int kb0) {
    int row = t >> 3, ch = t & 7;                 // K: 32 rows x 8 chunks
    async_ld_b128(lds_off(&sK[buf][row * 72 + ch * 8]),
                  &kptr[(kb0 + row) * HEAD_DIM + ch * 8]);
    int dd = t >> 2, kc = t & 3;                  // V^T: 64 d-rows x 4 chunks
    async_ld_b128(lds_off(&sVT[buf][dd * 40 + kc * 8]),
                  &vtptr[(size_t)dd * SEQ + kb0 + kc * 8]);
  };

  // Q^T B-operands resident in registers: lane n = query, VGPR v holds
  // d = half*16 + 2v (+1);  bq0 covers d 0..31, bq1 covers d 32..63.
  ABOp bq0, bq1;
#pragma unroll
  for (int v = 0; v < 8; ++v) {
    int d0 = half * 16 + v * 2;
    bq0.u[v] = *(const u32*)&qptr[(qrow + l) * HEAD_DIM + d0];
    bq1.u[v] = *(const u32*)&qptr[(qrow + l) * HEAD_DIM + 32 + d0];
  }

  float m_run = -3.0e38f, l_run = 0.f;
  f32x8 acc[4];
#pragma unroll
  for (int i = 0; i < 4; ++i) acc[i] = zero8();

  issue_tile(0, 0);
  int buf = 0;
  for (int kb0 = 0; kb0 < SEQ; kb0 += 32) {
    wait_async0();          // this wave's DMA done (tile `buf` complete)
    __syncthreads();        // all waves' DMA done; prev compute done
    if (kb0 + 32 < SEQ) issue_tile(buf ^ 1, kb0 + 32);   // prefetch next

    // S^T = K * Q^T : two 16-key subtiles, accumulate over d (2 chunks of 32)
    f32x8 s0 = zero8(), s1 = zero8();
#pragma unroll
    for (int c = 0; c < 2; ++c) {
      ABOp a0, a1;
#pragma unroll
      for (int v = 0; v < 8; ++v) {
        int k0 = half * 16 + v * 2 + c * 32;
        a0.u[v] = *(const u32*)&sK[buf][(l     ) * 72 + k0];
        a1.u[v] = *(const u32*)&sK[buf][(16 + l) * 72 + k0];
      }
      const ABOp bqc = c ? bq1 : bq0;
      s0 = wmma_bf16(a0, bqc, s0);
      s1 = wmma_bf16(a1, bqc, s1);
    }

    // online softmax: lanes = queries -> per-lane reductions + one xor-16
    float pmax = s0[0];
#pragma unroll
    for (int r = 1; r < 8; ++r) pmax = fmaxf(pmax, s0[r]);
#pragma unroll
    for (int r = 0; r < 8; ++r) pmax = fmaxf(pmax, s1[r]);
    pmax = fmaxf(pmax, __shfl_xor(pmax, 16, 32));
    float mnew = fmaxf(m_run, pmax);
    float rs = exp2f(m_run - mnew);
#pragma unroll
    for (int r = 0; r < 8; ++r) { s0[r] = exp2f(s0[r] - mnew); s1[r] = exp2f(s1[r] - mnew); }
    float ps = 0.f;
#pragma unroll
    for (int r = 0; r < 8; ++r) ps += s0[r] + s1[r];
    ps += __shfl_xor(ps, 16, 32);
    l_run = l_run * rs + ps;
    m_run = mnew;
#pragma unroll
    for (int tt = 0; tt < 4; ++tt)
#pragma unroll
      for (int r = 0; r < 8; ++r) acc[tt][r] *= rs;

    // P^T B-operand: other half's key range via one xor-16 shuffle, pack bf16
    float o0[8], o1[8];
#pragma unroll
    for (int r = 0; r < 8; ++r) {
      o0[r] = __shfl_xor(s0[r], 16, 32);
      o1[r] = __shfl_xor(s1[r], 16, 32);
    }
    ABOp bp;
#pragma unroll
    for (int v = 0; v < 8; ++v) {
      int j0 = 2 * v, j1 = 2 * v + 1;
      float x0 = half ? (j0 < 8 ? o1[j0] : s1[j0 - 8])
                      : (j0 < 8 ? s0[j0] : o0[j0 - 8]);
      float x1 = half ? (j1 < 8 ? o1[j1] : s1[j1 - 8])
                      : (j1 < 8 ? s0[j1] : o0[j1 - 8]);
      bp.u[v] = pkbf(x0, x1);
    }

    // O^T += V^T * P^T  (4 d-subtiles)
#pragma unroll
    for (int tt = 0; tt < 4; ++tt) {
      ABOp a;
#pragma unroll
      for (int v = 0; v < 8; ++v) {
        int k0 = half * 16 + v * 2;
        a.u[v] = *(const u32*)&sVT[buf][(tt * 16 + l) * 40 + k0];
      }
      acc[tt] = wmma_bf16(a, bp, acc[tt]);
    }
    buf ^= 1;
  }

  // normalize and store bf16 [b, h, n, d] (== row-major [8192,1024] for GEMM)
  float inv = 1.0f / l_run;
  u16* optr = og + (size_t)bh * SEQ * HEAD_DIM;
#pragma unroll
  for (int tt = 0; tt < 4; ++tt)
#pragma unroll
    for (int r = 0; r < 8; ++r) {
      int d = tt * 16 + r + 8 * half;
      optr[(qrow + l) * HEAD_DIM + d] = f2bf(acc[tt][r] * inv);
    }
}

// ---------------------------------------------------------------------------
// Output projection: out = attn_bf16 * Wo^T + bo, fp32 result.
// A tile (already bf16) staged via async-to-LDS DMA; W converted in VALU.
// ---------------------------------------------------------------------------
__global__ __launch_bounds__(256)
void out_proj_kernel(const u16* __restrict__ Xbf, const float* __restrict__ Wo,
                     const float* __restrict__ bo, float* __restrict__ out)
{
  __shared__ __align__(16) u16 sA[128 * 40];
  __shared__ __align__(16) u16 sB[64 * 40];

  const int t    = threadIdx.x;
  const int wave = t >> 5;
  const int lane = t & 31;
  const int half = lane >> 4;
  const int l    = lane & 15;
  const int mbase = blockIdx.x * 128;
  const int cbase = blockIdx.y * 64;

  f32x8 acc[4];
#pragma unroll
  for (int i = 0; i < 4; ++i) acc[i] = zero8();

  for (int kb = 0; kb < DIM; kb += 32) {
    // A tile 128x32 bf16: pure DMA, no VGPR data
#pragma unroll
    for (int i = 0; i < 2; ++i) {
      int c = t + i * 256;                       // 512 x 16B chunks
      int row = c >> 2, ch = c & 3;
      async_ld_b128(lds_off(&sA[row * 40 + ch * 8]),
                    &Xbf[(size_t)(mbase + row) * DIM + kb + ch * 8]);
    }
    // W tile fp32 -> bf16 (needs conversion, so through VGPRs)
#pragma unroll
    for (int i = 0; i < 2; ++i) {
      int c = t + i * 256;
      int row = c >> 3, f4 = c & 7;
      float4 v = *(const float4*)&Wo[(cbase + row) * DIM + kb + f4 * 4];
      *(uint2*)&sB[row * 40 + f4 * 4] = make_uint2(pkbf(v.x, v.y), pkbf(v.z, v.w));
    }
    wait_async0();
    __syncthreads();

    ABOp a;
#pragma unroll
    for (int v = 0; v < 8; ++v) {
      int k0 = half * 16 + v * 2;
      a.u[v] = *(const u32*)&sA[(wave * 16 + l) * 40 + k0];
    }
#pragma unroll
    for (int nt = 0; nt < 4; ++nt) {
      ABOp b;
#pragma unroll
      for (int v = 0; v < 8; ++v)
        b.u[v] = *(const u32*)&sB[(nt * 16 + l) * 40 + half * 16 + v * 2];
      acc[nt] = wmma_bf16(a, b, acc[nt]);
    }
    __syncthreads();
  }

#pragma unroll
  for (int nt = 0; nt < 4; ++nt) {
    int cg = cbase + nt * 16 + l;
    float bsv = bo[cg];
#pragma unroll
    for (int r = 0; r < 8; ++r) {
      int grow = mbase + wave * 16 + r + 8 * half;
      out[(size_t)grow * DIM + cg] = acc[nt][r] + bsv;
    }
  }
}

// ---------------------------------------------------------------------------
extern "C" void kernel_launch(void* const* d_in, const int* in_sizes, int n_in,
                              void* d_out, int out_size, void* d_ws, size_t ws_size,
                              hipStream_t stream)
{
  const float* query = (const float*)d_in[0];
  const float* key   = (const float*)d_in[1];
  const float* value = (const float*)d_in[2];
  const float* Wq    = (const float*)d_in[3];
  const float* bq    = (const float*)d_in[4];
  const float* Wk    = (const float*)d_in[5];
  const float* bk    = (const float*)d_in[6];
  const float* Wv    = (const float*)d_in[7];
  const float* bv    = (const float*)d_in[8];
  const float* Wo    = (const float*)d_in[9];
  const float* bo    = (const float*)d_in[10];
  float* out = (float*)d_out;

  u16* ws0 = (u16*)d_ws;
  const size_t T = (size_t)ROWS * DIM;       // 8M bf16 per tensor
  u16* q_bf  = ws0;
  u16* k_bf  = ws0 + T;
  u16* vt_bf = ws0 + 2 * T;                  // [b,h,d,n]
  u16* a_bf  = ws0 + 3 * T;                  // total 64 MB workspace

  proj_qkv_kernel<<<dim3(ROWS / 128, DIM / 64, 3), 256, 0, stream>>>(
      query, key, value, Wq, Wk, Wv, bq, bk, bv, q_bf, k_bf, vt_bf);
  flash_attn_kernel<<<dim3(SEQ / 128, BATCH * NHEADS), 256, 0, stream>>>(
      q_bf, k_bf, vt_bf, a_bf);
  out_proj_kernel<<<dim3(ROWS / 128, DIM / 64), 256, 0, stream>>>(
      a_bf, Wo, bo, out);
}
